// RelationalGraphLayer_50019189129705
// MI455X (gfx1250) — matrix operand order repaired and checked by reading, not compiled, not run on hardware
//
#include <hip/hip_runtime.h>

typedef __attribute__((ext_vector_type(16))) __bf16        v16bf;
typedef __attribute__((ext_vector_type(8)))  float         v8f;
typedef __attribute__((ext_vector_type(4)))  unsigned int  v4u;
typedef __attribute__((ext_vector_type(8)))  int           v8i32;
typedef __attribute__((ext_vector_type(4)))  int           v4i32;

#define B_NODES 4096
#define U_NBR   16384
#define D_FEAT  128
#define KSEG    4                      // K split across blocks
#define KSEG_LEN (U_NBR / KSEG)        // 4096 K per block
#define KC      64                     // K-chunk staged through LDS (2 k-steps)
#define CHUNK_SHORTS (KC * D_FEAT)     // 8192 bf16 = 16 KB per chunk

#if __has_builtin(__builtin_amdgcn_tensor_load_to_lds) && __has_builtin(__builtin_amdgcn_s_wait_tensorcnt)
#define USE_TDM 1
#else
#define USE_TDM 0
#endif

#if __has_builtin(__builtin_nontemporal_load)
#define NT_LOAD(p) __builtin_nontemporal_load(p)
#else
#define NT_LOAD(p) (*(p))
#endif

static __device__ __forceinline__ unsigned int f32_to_bf16_rne(float f) {
    unsigned int u = __float_as_uint(f);
    return (u + 0x7FFFu + ((u >> 16) & 1u)) >> 16;
}

static __device__ __forceinline__ unsigned int pack_mask_pair(unsigned int lo, unsigned int hi) {
    return lo * 0x3F80u + hi * 0x3F800000u;   // bf16(1.0)=0x3F80
}

#if USE_TDM
// 1D TDM: copy CHUNK_SHORTS bf16 (16 KB) from global to LDS.
// D# per CDNA5 ISA: count=1, type=2 (image),
// data_size=1 (2 bytes), tensor_dim0 = tile_dim0 = 8192, tensor_dim1 = tile_dim1 = 1.
// 6-arg builtin on this toolchain: (g0, g1, g2, g3, g_ext, cpol).
static __device__ __forceinline__ void tdm_load_chunk(unsigned lds_byte_addr,
                                                      const unsigned short* gsrc) {
    unsigned long long ga = (unsigned long long)(const void*)gsrc;
    v4u g0;
    g0.x = 1u;                                           // count=1, user descriptor
    g0.y = lds_byte_addr;                                // lds_addr
    g0.z = (unsigned)(ga & 0xFFFFFFFFu);                 // global_addr[31:0]
    g0.w = (unsigned)((ga >> 32) & 0x1FFFFFFu) | (2u << 30); // global_addr[56:32] | type=2
    v8i32 g1;
    g1[0] = 1 << 16;                                     // wg_mask=0, data_size=2B
    g1[1] = (int)((CHUNK_SHORTS & 0xFFFF) << 16);        // tensor_dim0 lo16 @ bits[63:48]
    g1[2] = (int)(CHUNK_SHORTS >> 16) | (1 << 16);       // tensor_dim0 hi16 | tensor_dim1 lo16=1
    g1[3] = (int)((CHUNK_SHORTS & 0xFFFF) << 16);        // tensor_dim1 hi16=0 | tile_dim0 @ [127:112]
    g1[4] = 1;                                           // tile_dim1=1, tile_dim2=0
    g1[5] = CHUNK_SHORTS;                                // tensor_dim0_stride lo32
    g1[6] = 0;
    g1[7] = 0;
    v4i32 z4 = {0, 0, 0, 0};
    v8i32 z8 = {0, 0, 0, 0, 0, 0, 0, 0};
    __builtin_amdgcn_tensor_load_to_lds(g0, g1, z4, z4, z8, 0);
}
#endif

// ---------------------------------------------------------------------------
// Kernel 1: gather table rows by idx, f32 -> bf16, stored in WMMA B-fragment
// layout; tile order [rel][kt(512)][nt(8)][512 bf16] so any K-chunk is a
// contiguous block (TDM/LDS friendly).
// ---------------------------------------------------------------------------
__global__ void __launch_bounds__(256)
gather_swizzle_kernel(const int* __restrict__ idx_code,
                      const int* __restrict__ idx_report,
                      const float* __restrict__ C_table,
                      const float* __restrict__ R_table,
                      unsigned short* __restrict__ Ebf) {
    int t    = blockIdx.x * blockDim.x + threadIdx.x;   // 2*512*8*32 threads
    int lane = t & 31;
    int nt   = (t >> 5) & 7;
    int kt   = (t >> 8) & 511;
    int rel  = t >> 17;

    const int*   idx = rel ? idx_report : idx_code;
    const float* tab = rel ? R_table    : C_table;

    int n     = nt * 16 + (lane & 15);
    int kbase = kt * 32 + ((lane < 16) ? 0 : 8);

    unsigned int w[8];
#pragma unroll
    for (int j = 0; j < 8; ++j) {
        int u0 = kbase + 2 * j + ((j >= 4) ? 8 : 0);
        float f0 = tab[(size_t)idx[u0]     * D_FEAT + n];
        float f1 = tab[(size_t)idx[u0 + 1] * D_FEAT + n];
        w[j] = f32_to_bf16_rne(f0) | (f32_to_bf16_rne(f1) << 16);
    }
    size_t base = ((((size_t)rel * 512 + kt) * 8 + nt) * 512) + (size_t)lane * 16;
    uint4* dst = (uint4*)(Ebf + base);
    dst[0] = make_uint4(w[0], w[1], w[2], w[3]);
    dst[1] = make_uint4(w[4], w[5], w[6], w[7]);
}

// ---------------------------------------------------------------------------
// Kernel 2: per (rel, kseg) partial GEMM. Wave owns a 16-row M-tile x 128 N.
// deg computed by a 9th WMMA vs an all-ones B tile (lands in C layout, so
// scaling needs no cross-lane traffic). E slabs double-buffered in LDS via TDM.
// B fragments consumed through a distance-2 rotating pipeline so two
// ds_load_b128 pairs stay in flight over every WMMA.
// Writes non-atomic partials S[rel][seg][4096][128] and deg[rel][seg][4096].
// ---------------------------------------------------------------------------
__global__ void __launch_bounds__(256)
rel_gemm_kernel(const int* __restrict__ mask_code,
                const int* __restrict__ mask_report,
                const unsigned short* __restrict__ Ebf,
                float* __restrict__ S,
                float* __restrict__ degp) {
    __shared__ __align__(16) unsigned short smem[2 * CHUNK_SHORTS];  // 32 KB

    const int tid  = threadIdx.x;
    const int lane = tid & 31;
    const int wave = tid >> 5;
    const int seg  = blockIdx.y;
    const int rel  = blockIdx.z;
    const int m0   = blockIdx.x * 128 + wave * 16;
    const int kbeg = seg * KSEG_LEN;

    const int* mask          = rel ? mask_report : mask_code;
    const unsigned short* Eb = Ebf + (size_t)rel * U_NBR * D_FEAT;
    const unsigned short* Eseg = Eb + (size_t)(kbeg >> 5) * (8 * 512);
    const int* mrow = mask + (size_t)(m0 + (lane & 15)) * U_NBR + kbeg;
    const int abase = (lane < 16) ? 0 : 8;

    // all-ones bf16 B tile for the degree WMMA
    union { unsigned int u[8]; v16bf v; } Bone;
#pragma unroll
    for (int j = 0; j < 8; ++j) Bone.u[j] = 0x3F803F80u;

    v8f acc[8], dacc;
#pragma unroll
    for (int nt = 0; nt < 8; ++nt)
        acc[nt] = (v8f){0.f, 0.f, 0.f, 0.f, 0.f, 0.f, 0.f, 0.f};
    dacc = (v8f){0.f, 0.f, 0.f, 0.f, 0.f, 0.f, 0.f, 0.f};

    const int nchunks = KSEG_LEN / KC;   // 64

#if USE_TDM
    const unsigned lds0 = (unsigned)(uintptr_t)(void*)smem;
    if (wave == 0) tdm_load_chunk(lds0, Eseg);
#endif

    for (int c = 0; c < nchunks; ++c) {
        const int buf = c & 1;
#if USE_TDM
        if (wave == 0) {
            __builtin_amdgcn_s_wait_tensorcnt(0);   // chunk c resident
        }
        __syncthreads();                             // broadcast completion; frees other buf
        if (wave == 0 && (c + 1) < nchunks) {
            tdm_load_chunk(lds0 + (unsigned)((c + 1) & 1) * (CHUNK_SHORTS * 2),
                           Eseg + (size_t)(c + 1) * CHUNK_SHORTS);
        }
#else
        __syncthreads();
        {   // fallback: cooperative register staging (alternates the two buffers, no overlap)
            const uint4* src = (const uint4*)(Eseg + (size_t)c * CHUNK_SHORTS);
            uint4* dst = (uint4*)(smem + buf * CHUNK_SHORTS);
#pragma unroll
            for (int i = 0; i < 4; ++i) dst[tid + i * 256] = src[tid + i * 256];
        }
        __syncthreads();
#endif
        const unsigned short* sbuf = smem + buf * CHUNK_SHORTS;

#pragma unroll
        for (int kk = 0; kk < KC / 32; ++kk) {
            const int k0 = c * KC + kk * 32;
            // ---- streamed (single-use) mask loads: non-temporal ----
            const v4u a0 = NT_LOAD((const v4u*)(mrow + k0 + abase));
            const v4u a1 = NT_LOAD((const v4u*)(mrow + k0 + abase + 4));
            const v4u a2 = NT_LOAD((const v4u*)(mrow + k0 + abase + 16));
            const v4u a3 = NT_LOAD((const v4u*)(mrow + k0 + abase + 20));
            union { unsigned int u[8]; v16bf v; } A;
            A.u[0] = pack_mask_pair(a0.x, a0.y);
            A.u[1] = pack_mask_pair(a0.z, a0.w);
            A.u[2] = pack_mask_pair(a1.x, a1.y);
            A.u[3] = pack_mask_pair(a1.z, a1.w);
            A.u[4] = pack_mask_pair(a2.x, a2.y);
            A.u[5] = pack_mask_pair(a2.z, a2.w);
            A.u[6] = pack_mask_pair(a3.x, a3.y);
            A.u[7] = pack_mask_pair(a3.z, a3.w);

            const unsigned short* sb = sbuf + kk * (8 * 512) + lane * 16;

            // ---- distance-2 rotating pipeline over the 8 B fragments ----
            v16bf b0 = *(const v16bf*)(sb);
            v16bf b1 = *(const v16bf*)(sb + 512);

            // degree WMMA first: no LDS dependency, runs while b0/b1 land
            dacc = __builtin_amdgcn_wmma_f32_16x16x32_bf16(
                false, A.v, false, Bone.v, (short)0, dacc, false, false);

#pragma unroll
            for (int nt = 0; nt < 8; ++nt) {
                v16bf b2 = (nt < 6) ? *(const v16bf*)(sb + (nt + 2) * 512) : b1;
                acc[nt] = __builtin_amdgcn_wmma_f32_16x16x32_bf16(
                    false, A.v, false, b0, (short)0, acc[nt], false, false);
                b0 = b1;
                b1 = b2;
            }
        }
    }

    // ---- write per-segment partials (each element owned by one thread) ----
    float* Sseg = S + (size_t)(rel * KSEG + seg) * B_NODES * D_FEAT;
#pragma unroll
    for (int nt = 0; nt < 8; ++nt) {
#pragma unroll
        for (int j = 0; j < 8; ++j) {
            const int rl = j + ((lane < 16) ? 0 : 8);
            Sseg[(size_t)(m0 + rl) * D_FEAT + nt * 16 + (lane & 15)] = acc[nt][j];
        }
    }
    if ((lane & 15) == 0) {
        float* dseg = degp + (size_t)(rel * KSEG + seg) * B_NODES;
#pragma unroll
        for (int j = 0; j < 8; ++j) {
            const int rl = j + ((lane < 16) ? 0 : 8);
            dseg[m0 + rl] = dacc[j];
        }
    }
}

// ---------------------------------------------------------------------------
// Kernel 3: reduce K-segments, divide by degree, gamma-combine.
// ---------------------------------------------------------------------------
__global__ void __launch_bounds__(256)
finalize_kernel(const float* __restrict__ S,
                const float* __restrict__ degp,
                const float* __restrict__ gamma,
                float* __restrict__ out) {
    const int t   = blockIdx.x * blockDim.x + threadIdx.x;  // B*D threads
    const int row = t >> 7;
    const float g = gamma[0];

    float s0 = 0.f, s1 = 0.f, d0 = 0.f, d1 = 0.f;
#pragma unroll
    for (int seg = 0; seg < KSEG; ++seg) {
        s0 += S[(size_t)(0 * KSEG + seg) * B_NODES * D_FEAT + t];
        s1 += S[(size_t)(1 * KSEG + seg) * B_NODES * D_FEAT + t];
        d0 += degp[(size_t)(0 * KSEG + seg) * B_NODES + row];
        d1 += degp[(size_t)(1 * KSEG + seg) * B_NODES + row];
    }
    const float r0 = (d0 > 0.f) ? s0 / d0 : 0.f;
    const float r1 = (d1 > 0.f) ? s1 / d1 : 0.f;
    out[t] = g * r0 + (1.f - g) * r1;
}

extern "C" void kernel_launch(void* const* d_in, const int* in_sizes, int n_in,
                              void* d_out, int out_size, void* d_ws, size_t ws_size,
                              hipStream_t stream) {
    const int*   mask_code   = (const int*)d_in[0];
    const int*   idx_code    = (const int*)d_in[1];
    const int*   mask_report = (const int*)d_in[2];
    const int*   idx_report  = (const int*)d_in[3];
    const float* C_table     = (const float*)d_in[4];
    const float* R_table     = (const float*)d_in[5];
    const float* gamma       = (const float*)d_in[6];
    float* out = (float*)d_out;

    // ws layout: [0,8MB) Ebf  | [8MB,24MB) S partials | [24MB,+128KB) degrees
    unsigned short* Ebf  = (unsigned short*)d_ws;
    float*          S    = (float*)((char*)d_ws + (8u << 20));
    float*          degp = (float*)((char*)d_ws + (24u << 20));

    gather_swizzle_kernel<<<1024, 256, 0, stream>>>(idx_code, idx_report,
                                                    C_table, R_table, Ebf);
    dim3 grid(B_NODES / 128, KSEG, 2);          // 32 x 4 x 2 = 256 blocks
    rel_gemm_kernel<<<grid, 256, 0, stream>>>(mask_code, mask_report, Ebf, S, degp);

    finalize_kernel<<<(B_NODES * D_FEAT) / 256, 256, 0, stream>>>(S, degp, gamma, out);
}
